// AutoRegressiveDistribution_2551210574396
// MI455X (gfx1250) — matrix operand order repaired and checked by reading, not compiled, not run on hardware
//
#include <hip/hip_runtime.h>
#include <hip/hip_fp16.h>

typedef _Float16 f16;
typedef __attribute__((ext_vector_type(16))) _Float16 v16h;
typedef __attribute__((ext_vector_type(8)))  _Float16 v8h;
typedef __attribute__((ext_vector_type(8)))  float    v8f;
typedef __attribute__((ext_vector_type(4)))  float    v4f;

#define S_   2
#define B_   1024
#define D_   128
#define H_   512
#define CTX_ 128
#define HP   516   // padded LDS row stride in floats (stagger banks per row)

// 32-bit LDS byte offset of a __shared__ pointer (generic -> AS3 -> int)
__device__ __forceinline__ unsigned lds_off_u32(const void* p) {
    return (unsigned)(uintptr_t)(__attribute__((address_space(3))) const char*)p;
}

// one async global->LDS 128b copy per lane (GVS addressing; IOFFSET applies to
// both the LDS and global side, so chunking via offset advances both)
template <int IOFF>
__device__ __forceinline__ void async_b128(unsigned lds_dst, unsigned voff, const void* sbase) {
    asm volatile("global_load_async_to_lds_b128 %0, %1, %2 offset:%3"
                 :: "v"(lds_dst), "v"(voff), "s"(sbase), "n"(IOFF)
                 : "memory");
}

// ---------------------------------------------------------------------------
// Prep: f16 packs of context/Wc, plus degree-mask-folded weights:
//   W1tm[d][h]    = W1[h,d]      * ((h%127) >= d)   (f32, column-d contiguous)
//   W2md[d][j][h] = W2[j*D+d, h] * ((h%127) <  d)   (f16; j=0 mu row, j=1 pre row)
// ---------------------------------------------------------------------------
__global__ void prep_kernel(const float* __restrict__ context,
                            const float* __restrict__ W1,
                            const float* __restrict__ Wc,
                            const float* __restrict__ W2,
                            f16* __restrict__ ctxh, f16* __restrict__ Wch,
                            f16* __restrict__ W2md, float* __restrict__ W1tm) {
    int i = blockIdx.x * blockDim.x + threadIdx.x;
    if (i < B_ * CTX_) ctxh[i] = (f16)context[i];
    if (i < H_ * CTX_) Wch[i] = (f16)Wc[i];
    if (i < D_ * 2 * H_) {
        int h = i & (H_ - 1);
        int dj = i >> 9, j = dj & 1, d = dj >> 1;
        float v = W2[(j * D_ + d) * H_ + h];
        W2md[i] = ((h % 127) < d) ? (f16)v : (f16)0.0f;
    }
    if (i < D_ * H_) {
        int h = i & (H_ - 1), d = i >> 9;
        float v = W1[h * D_ + d];
        W1tm[i] = ((h % 127) >= d) ? v : 0.0f;
    }
}

// ---------------------------------------------------------------------------
// ctx_term(1024x512) = context(1024x128) @ Wc^T + b1, via v_wmma_f32_16x16x32_f16.
// One wave per 16x16 tile; grid = 64 (M) * 32 (N) = 2048 blocks of 32.
// ---------------------------------------------------------------------------
__global__ void __launch_bounds__(32)
ctx_gemm_kernel(const f16* __restrict__ ctxh, const f16* __restrict__ Wch,
                const float* __restrict__ b1, float* __restrict__ ctx_term) {
    const int lane = threadIdx.x;
    const int hid  = lane >> 4;
    const int l15  = lane & 15;
    const int tM = blockIdx.x & 63, tN = blockIdx.x >> 6;
    const int b0 = tM * 16, h0 = tN * 16;
    const int arow = b0 + l15;
    const int ncol = h0 + l15;
    v8f acc = {};
    for (int k0 = 0; k0 < CTX_; k0 += 32) {
        int ab = k0 + hid * 8;                    // A: K chunks [ab,ab+8) u [ab+16,ab+24)
        v8h a0 = *(const v8h*)(ctxh + arow * CTX_ + ab);
        v8h a1 = *(const v8h*)(ctxh + arow * CTX_ + ab + 16);
        int kb = k0 + hid * 8;                    // B: K chunks [kb,kb+8) u [kb+16,kb+24), N = l15
        v8h w0 = *(const v8h*)(Wch + ncol * CTX_ + kb);
        v8h w1 = *(const v8h*)(Wch + ncol * CTX_ + kb + 16);
        v16h A, Bm;
        for (int t = 0; t < 8; ++t) { A[t] = a0[t]; A[t + 8] = a1[t];
                                      Bm[t] = w0[t]; Bm[t + 8] = w1[t]; }
        acc = __builtin_amdgcn_wmma_f32_16x16x32_f16(false, A, false, Bm,
                                                     (short)0, acc, false, false);
    }
    const float bias = b1[ncol];
    for (int j = 0; j < 8; ++j)
        ctx_term[(b0 + j + hid * 8) * H_ + ncol] = acc[j] + bias;
}

// ---------------------------------------------------------------------------
// Persistent recursion: 128 blocks x 1 wave, 16 rows each; h_pre in LDS.
// Per step: WMMA out(16x16) = relu(h_pre) @ [W2md[d][0],W2md[d][1],0..];
// z_d = mu + softplus(pre)*eps;  h_pre += z_d (outer) W1tm[d].
// Next step's 4KB of weights are double-buffered into LDS with
// global_load_async_to_lds_b128, overlapped with current step's compute.
// ---------------------------------------------------------------------------
__global__ void __launch_bounds__(32)
made_recursion_kernel(const float* __restrict__ ctx_term,
                      const float* __restrict__ eps,
                      const float* __restrict__ b2,
                      const f16*   __restrict__ W2md,
                      const float* __restrict__ W1tm,
                      float*       __restrict__ zout) {
    __shared__ float sH[16 * HP];
    __shared__ f16   sW2[2][2 * H_];   // double-buffered W2md[d] (2KB each)
    __shared__ float sW1[2][H_];       // double-buffered W1tm[d] (2KB each)
    __shared__ float sMu[16];
    __shared__ float sPre[16];
    __shared__ float sZ[16];

    const int lane = threadIdx.x;
    const int hid  = lane >> 4;
    const int m    = lane & 15;
    const int r0   = blockIdx.x * 16;

    const unsigned sW2base = lds_off_u32(&sW2[0][0]);
    const unsigned sW1base = lds_off_u32(&sW1[0][0]);
    const unsigned l16 = (unsigned)lane * 16u;

    // kick off async staging of step-0 weights (8 x 512B)
    {
        unsigned dW2 = sW2base + l16, dW1 = sW1base + l16;
        async_b128<0>(dW2, l16, W2md);   async_b128<512>(dW2, l16, W2md);
        async_b128<1024>(dW2, l16, W2md); async_b128<1536>(dW2, l16, W2md);
        async_b128<0>(dW1, l16, W1tm);   async_b128<512>(dW1, l16, W1tm);
        async_b128<1024>(dW1, l16, W1tm); async_b128<1536>(dW1, l16, W1tm);
    }

    // init h_pre from ctx_term (b = row mod B; S=2 repeats context)
    for (int i = lane; i < 16 * (H_ / 4); i += 32) {
        int mm = i >> 7;
        int hh = (i & 127) * 4;
        int b = (r0 + mm) & (B_ - 1);
        *(v4f*)(sH + mm * HP + hh) = *(const v4f*)(ctx_term + b * H_ + hh);
    }
    __syncthreads();

    for (int d = 0; d < D_; ++d) {
        const int buf = d & 1;
        // prefetch step d+1 into the other buffer
        if (d + 1 < D_) {
            unsigned voff = (unsigned)(d + 1) * 2048u + l16;   // 2KB per step in both arrays
            unsigned dW2 = sW2base + (unsigned)((d + 1) & 1) * 4096u + l16;
            unsigned dW1 = sW1base + (unsigned)((d + 1) & 1) * 2048u + l16;
            async_b128<0>(dW2, voff, W2md);   async_b128<512>(dW2, voff, W2md);
            async_b128<1024>(dW2, voff, W2md); async_b128<1536>(dW2, voff, W2md);
            async_b128<0>(dW1, voff, W1tm);   async_b128<512>(dW1, voff, W1tm);
            async_b128<1024>(dW1, voff, W1tm); async_b128<1536>(dW1, voff, W1tm);
            asm volatile("s_wait_asynccnt 0x8" ::: "memory");  // step-d buffers ready
        } else {
            asm volatile("s_wait_asynccnt 0x0" ::: "memory");
        }

        v8f acc = {};
        const f16* w2l = &sW2[buf][0] + m * H_;          // rows j=0 (mu), j=1 (pre); valid m<2
        for (int k0 = 0; k0 < H_; k0 += 32) {
            const int ab = k0 + hid * 8;
            const float* hp = sH + m * HP;
            v4f p0 = *(const v4f*)(hp + ab);
            v4f p1 = *(const v4f*)(hp + ab + 4);
            v4f q0 = *(const v4f*)(hp + ab + 16);
            v4f q1 = *(const v4f*)(hp + ab + 20);
            v16h A;
            for (int t = 0; t < 4; ++t) {
                A[t]      = (f16)fmaxf(p0[t], 0.0f);
                A[t + 4]  = (f16)fmaxf(p1[t], 0.0f);
                A[t + 8]  = (f16)fmaxf(q0[t], 0.0f);
                A[t + 12] = (f16)fmaxf(q1[t], 0.0f);
            }
            v16h Bm = {};
            if (m < 2) {
                int kb = k0 + hid * 8;
                v8h w0 = *(const v8h*)(w2l + kb);
                v8h w1 = *(const v8h*)(w2l + kb + 16);
                for (int t = 0; t < 8; ++t) { Bm[t] = w0[t]; Bm[t + 8] = w1[t]; }
            }
            acc = __builtin_amdgcn_wmma_f32_16x16x32_f16(false, A, false, Bm,
                                                         (short)0, acc, false, false);
        }
        // scatter the two useful C columns (N=0: mu, N=1: pre) to LDS
        if (m < 2) {
            float* dst = (m == 0) ? sMu : sPre;
            for (int j = 0; j < 8; ++j) dst[hid * 8 + j] = acc[j];
        }
        __syncthreads();
        if (lane < 16) {
            const int r = r0 + lane;
            float mu  = sMu[lane]  + b2[d];
            float pre = sPre[lane] + b2[d + D_];
            float sp;                               // stable softplus
            if      (pre >  20.0f) sp = pre;
            else if (pre < -20.0f) sp = expf(pre);
            else                   sp = log1pf(expf(pre));
            float z = mu + sp * eps[r * D_ + d];
            sZ[lane] = z;
            zout[r * D_ + d] = z;
        }
        __syncthreads();
        // rank-1 update: h_pre[mm][:] += z[mm] * W1tm[d][:], skipping all-zero chunks
        const float* wc = &sW1[buf][0];
        v4f w[4];
        bool nz[4];
        for (int j = 0; j < 4; ++j) {
            w[j] = *(const v4f*)(wc + 4 * (lane + 32 * j));
            nz[j] = (w[j][0] != 0.0f) | (w[j][1] != 0.0f) |
                    (w[j][2] != 0.0f) | (w[j][3] != 0.0f);
        }
        for (int mm = 0; mm < 16; ++mm) {
            const float zm = sZ[mm];
            float* hp = sH + mm * HP;
            for (int j = 0; j < 4; ++j) {
                if (nz[j]) {
                    const int h = 4 * (lane + 32 * j);
                    v4f hv = *(v4f*)(hp + h);
                    hv += zm * w[j];
                    *(v4f*)(hp + h) = hv;
                }
            }
        }
        __syncthreads();
    }
}

// ---------------------------------------------------------------------------
extern "C" void kernel_launch(void* const* d_in, const int* in_sizes, int n_in,
                              void* d_out, int out_size, void* d_ws, size_t ws_size,
                              hipStream_t stream) {
    const float* context = (const float*)d_in[0];
    const float* eps     = (const float*)d_in[1];
    const float* W1      = (const float*)d_in[2];
    const float* Wc      = (const float*)d_in[3];
    const float* b1      = (const float*)d_in[4];
    const float* W2      = (const float*)d_in[5];
    const float* b2      = (const float*)d_in[6];
    float* zout = (float*)d_out;

    char* ws = (char*)d_ws;
    f16*   ctxh = (f16*)(ws);                       // 1024*128 f16   = 262144 B
    f16*   Wch  = (f16*)(ws + 262144);              //  512*128 f16   = 131072 B
    f16*   W2md = (f16*)(ws + 393216);              // 128*2*512 f16  = 262144 B
    float* W1tm = (float*)(ws + 655360);            // 128*512 f32    = 262144 B
    float* ctxT = (float*)(ws + 917504);            // 1024*512 f32   = 2097152 B
                                                    // total 3014656 B

    prep_kernel<<<512, 256, 0, stream>>>(context, W1, Wc, W2, ctxh, Wch, W2md, W1tm);
    ctx_gemm_kernel<<<2048, 32, 0, stream>>>(ctxh, Wch, b1, ctxT);
    made_recursion_kernel<<<128, 32, 0, stream>>>(ctxT, eps, b2, W2md, W1tm, zout);
}
